// simpleGNN_MR_59347858096284
// MI455X (gfx1250) — matrix operand order repaired and checked by reading, not compiled, not run on hardware
//
#include <hip/hip_runtime.h>
#include <hip/hip_bf16.h>

// ---------------------------------------------------------------------------
// simpleGNN on MI455X (gfx1250): all GEMMs via v_wmma_f32_16x16x32_f16.
//   - weights pre-packed on device into WMMA B-fragment order
//   - edge-MLP A operand gathered from an f16 copy of h (L2-resident, 10MB)
//   - edge-MLP processes 32 edges/wave (2 M-tiles) so every B fragment
//     feeds two WMMAs (halves weight-fragment traffic in the hot kernel)
//   - segment_sum via global_atomic_add_f32 into L2-resident agg
//   - branchless SELU (no exec-mask divergence)
//   - LDS re-layout buffers padded (+8 halves/row) to avoid bank conflicts
// Param input order assumes jax pytree (sorted-key) flattening:
//   [0]=feature, [1..12]=l{0,1}_e{0,1,2}_{src,dst},
//   per layer l (base=13+18l): Wout, bn.beta, bn.gamma, bn.mean, bn.var, bout,
//       em{e}: W1,W2,b1,b2  (e=0..2)
//   [49..52]=proj_in W1,W2,b1,b2   [53..56]=proj_out W1,W2,b1,b2
// ---------------------------------------------------------------------------

#define N0_  40000
#define N1_  20000
#define N2_  10000
#define E0_  160000
#define E1_  80000
#define INF_ 64
#define HID_ 128
#define MLP3_ 64

#define LDSP_H (HID_ + 8)    // padded LDS row stride (halves), proj_in
#define LDSP_E (MLP3_ + 8)   // padded LDS row stride (halves), edge kernel

typedef __attribute__((ext_vector_type(16))) _Float16 v16h;
typedef __attribute__((ext_vector_type(8)))  float    v8f;

__device__ __forceinline__ v8f wmma16(v16h a, v16h b, v8f c) {
  return __builtin_amdgcn_wmma_f32_16x16x32_f16(
      /*neg_a=*/false, a, /*neg_b=*/false, b,
      /*c_mod=*/(short)0, c, /*reuse_a=*/false, /*reuse_b=*/false);
}

// Branchless SELU: S*max(x,0) + S*A*(exp(min(x,0))-1); exp(0)-1==0 for x>=0.
__device__ __forceinline__ float selu_f(float x) {
  const float S  = 1.0507009873554805f;
  const float SA = 1.7580993408473766f;  // S*A
  float neg = __expf(fminf(x, 0.f)) - 1.f;
  return S * fmaxf(x, 0.f) + SA * neg;
}

// B fragment: packed[((nt*nK + ks)*32 + lane)*16 + j]
__device__ __forceinline__ v16h ldb(const _Float16* p0, int nt, int nK, int ks, int lane) {
  const _Float16* p = p0 + ((size_t)((nt * nK + ks) * 32 + lane) << 4);
  return *(const v16h*)p;
}

// A fragment gather: row-major source row; 16-bit A 16x32 layout:
//   m = lane&15 ; j<8 -> k = ks*32 + (lane>>4)*8 + j ; j>=8 -> +16
template <typename T>
__device__ __forceinline__ v16h gatherA(const T* row, int ks, int lane) {
  int kb = (ks << 5) + ((lane >> 4) << 3);
  v16h a;
#pragma unroll
  for (int i = 0; i < 8; i++) {
    a[i]     = (_Float16)row[kb + i];
    a[i + 8] = (_Float16)row[kb + 16 + i];
  }
  return a;
}

// ------------------------- weight packing ----------------------------------
__global__ void pack_b_kernel(const float* __restrict__ W, _Float16* __restrict__ outp,
                              int K, int N) {
  int idx = blockIdx.x * blockDim.x + threadIdx.x;
  if (idx >= K * N) return;
  int j    = idx & 15;
  int lane = (idx >> 4) & 31;
  int tt   = idx >> 9;
  int nK   = K >> 5;
  int ks   = tt % nK;
  int nt   = tt / nK;
  int n = nt * 16 + (lane & 15);
  int k = ks * 32 + ((lane >> 4) << 3) + (j & 7) + ((j >> 3) << 4);
  outp[idx] = (_Float16)W[(size_t)k * N + n];
}

__global__ void zero_kernel(float* p, int n) {
  int i = blockIdx.x * blockDim.x + threadIdx.x;
  if (i < n) p[i] = 0.f;
}

// ------------------------- proj_in: (N0,64)->selu->(N0,128)->selu ----------
__global__ __launch_bounds__(256) void proj_in_kernel(
    const float* __restrict__ feat,
    const _Float16* __restrict__ w1p, const float* __restrict__ b1,
    const _Float16* __restrict__ w2p, const float* __restrict__ b2,
    float* __restrict__ h32, _Float16* __restrict__ h16, float* __restrict__ inter0) {
  __shared__ _Float16 lds[8][16 * LDSP_H];
  const int lane = threadIdx.x & 31;
  const int wave = threadIdx.x >> 5;
  const int tile = blockIdx.x * 8 + wave;
  const int nTiles = N0_ / 16;
  const bool valid = tile < nTiles;
  const int t = valid ? tile : 0;
  const int m = lane & 15;
  const int node = t * 16 + m;
  const int mbase = (lane >> 4) << 3;

  const float* frow = feat + (size_t)node * INF_;
  v16h a0 = gatherA(frow, 0, lane);
  v16h a1 = gatherA(frow, 1, lane);

  _Float16* hl = &lds[wave][0];
#pragma unroll
  for (int nt = 0; nt < 8; nt++) {
    v8f c = {};
    c = wmma16(a0, ldb(w1p, nt, 2, 0, lane), c);
    c = wmma16(a1, ldb(w1p, nt, 2, 1, lane), c);
    int n = nt * 16 + m;
    float bias = b1[n];
#pragma unroll
    for (int v = 0; v < 8; v++)
      hl[(mbase + v) * LDSP_H + n] = (_Float16)selu_f(c[v] + bias);
  }
  __syncthreads();

  const _Float16* arow = &lds[wave][(size_t)m * LDSP_H];
  v16h A0 = gatherA(arow, 0, lane);
  v16h A1 = gatherA(arow, 1, lane);
  v16h A2 = gatherA(arow, 2, lane);
  v16h A3 = gatherA(arow, 3, lane);
#pragma unroll
  for (int nt = 0; nt < 8; nt++) {
    v8f c = {};
    c = wmma16(A0, ldb(w2p, nt, 4, 0, lane), c);
    c = wmma16(A1, ldb(w2p, nt, 4, 1, lane), c);
    c = wmma16(A2, ldb(w2p, nt, 4, 2, lane), c);
    c = wmma16(A3, ldb(w2p, nt, 4, 3, lane), c);
    int n = nt * 16 + m;
    float bias = b2[n];
    if (valid) {
#pragma unroll
      for (int v = 0; v < 8; v++) {
        int nm = t * 16 + mbase + v;
        float x = selu_f(c[v] + bias);
        h32[(size_t)nm * HID_ + n] = x;
        h16[(size_t)nm * HID_ + n] = (_Float16)x;
        if (nm < N2_) inter0[(size_t)nm * HID_ + n] = x;
      }
    }
  }
}

// ------ edge MLP: concat(h[s],h[d]) -> 64 -> 128, scatter-add --------------
// 32 edges per wave (two 16-row M-tiles); each B fragment feeds 2 WMMAs.
__global__ __launch_bounds__(256) void edge_mlp_kernel(
    const _Float16* __restrict__ h16,
    const int* __restrict__ src, const int* __restrict__ dst,
    const _Float16* __restrict__ w1p, const float* __restrict__ b1,
    const _Float16* __restrict__ w2p, const float* __restrict__ b2,
    float* __restrict__ agg, int nTiles32) {
  __shared__ _Float16 lds[8][32 * LDSP_E];
  __shared__ int dIdx[8][32];
  const int lane = threadIdx.x & 31;
  const int wave = threadIdx.x >> 5;
  const int tile = blockIdx.x * 8 + wave;
  const bool valid = tile < nTiles32;
  const int t = valid ? tile : 0;
  const int m = lane & 15;
  const int mbase = (lane >> 4) << 3;

  const int e0 = t * 32 + m;
  const int e1 = t * 32 + 16 + m;
  const int s0 = src[e0];
  const int d0 = dst[e0];
  const int s1 = src[e1];
  const int d1 = dst[e1];
  dIdx[wave][m] = d0;
  dIdx[wave][16 + m] = d1;

  const _Float16* s0row = h16 + (size_t)s0 * HID_;
  const _Float16* d0row = h16 + (size_t)d0 * HID_;
  const _Float16* s1row = h16 + (size_t)s1 * HID_;
  const _Float16* d1row = h16 + (size_t)d1 * HID_;
  v16h A0[8], A1[8];
#pragma unroll
  for (int ks = 0; ks < 4; ks++) {
    A0[ks]     = gatherA(s0row, ks, lane);
    A0[4 + ks] = gatherA(d0row, ks, lane);
    A1[ks]     = gatherA(s1row, ks, lane);
    A1[4 + ks] = gatherA(d1row, ks, lane);
  }

  _Float16* hl = &lds[wave][0];
#pragma unroll
  for (int nt = 0; nt < 4; nt++) {
    v8f c0 = {}, c1 = {};
#pragma unroll
    for (int ks = 0; ks < 8; ks++) {
      v16h b = ldb(w1p, nt, 8, ks, lane);
      c0 = wmma16(A0[ks], b, c0);
      c1 = wmma16(A1[ks], b, c1);
    }
    int n = nt * 16 + m;
    float bias = b1[n];
#pragma unroll
    for (int v = 0; v < 8; v++) {
      hl[(mbase + v) * LDSP_E + n]        = (_Float16)selu_f(c0[v] + bias);
      hl[(16 + mbase + v) * LDSP_E + n]   = (_Float16)selu_f(c1[v] + bias);
    }
  }
  __syncthreads();

  const _Float16* arow0 = &lds[wave][(size_t)m * LDSP_E];
  const _Float16* arow1 = &lds[wave][(size_t)(16 + m) * LDSP_E];
  v16h Ha0 = gatherA(arow0, 0, lane);
  v16h Ha1 = gatherA(arow0, 1, lane);
  v16h Hb0 = gatherA(arow1, 0, lane);
  v16h Hb1 = gatherA(arow1, 1, lane);
#pragma unroll
  for (int nt = 0; nt < 8; nt++) {
    v16h w0 = ldb(w2p, nt, 2, 0, lane);
    v16h w1 = ldb(w2p, nt, 2, 1, lane);
    v8f c0 = {}, c1 = {};
    c0 = wmma16(Ha0, w0, c0);
    c1 = wmma16(Hb0, w0, c1);
    c0 = wmma16(Ha1, w1, c0);
    c1 = wmma16(Hb1, w1, c1);
    int n = nt * 16 + m;
    float bias = b2[n];
    if (valid) {
#pragma unroll
      for (int v = 0; v < 8; v++) {
        int dd0 = dIdx[wave][mbase + v];
        int dd1 = dIdx[wave][16 + mbase + v];
        atomicAdd(&agg[(size_t)dd0 * HID_ + n], selu_f(c0[v] + bias));
        atomicAdd(&agg[(size_t)dd1 * HID_ + n], selu_f(c1[v] + bias));
      }
    }
  }
}

// ---------- node update: h = selu(bn(agg@Wout + bout + dst)) ---------------
__global__ __launch_bounds__(256) void node_update_kernel(
    const float* __restrict__ agg,
    const _Float16* __restrict__ woutp, const float* __restrict__ bout,
    const float* __restrict__ beta, const float* __restrict__ gamma,
    const float* __restrict__ mean, const float* __restrict__ var,
    float* __restrict__ h32, _Float16* __restrict__ h16,
    float* __restrict__ interBuf, int nTiles) {
  const int lane = threadIdx.x & 31;
  const int wave = threadIdx.x >> 5;
  const int tile = blockIdx.x * 8 + wave;
  const bool valid = tile < nTiles;
  const int t = valid ? tile : 0;
  const int m = lane & 15;
  const int mbase = (lane >> 4) << 3;
  const int node = t * 16 + m;

  const float* arow = agg + (size_t)node * HID_;
  v16h A0 = gatherA(arow, 0, lane);
  v16h A1 = gatherA(arow, 1, lane);
  v16h A2 = gatherA(arow, 2, lane);
  v16h A3 = gatherA(arow, 3, lane);
#pragma unroll
  for (int nt = 0; nt < 8; nt++) {
    v8f c = {};
    c = wmma16(A0, ldb(woutp, nt, 4, 0, lane), c);
    c = wmma16(A1, ldb(woutp, nt, 4, 1, lane), c);
    c = wmma16(A2, ldb(woutp, nt, 4, 2, lane), c);
    c = wmma16(A3, ldb(woutp, nt, 4, 3, lane), c);
    int n = nt * 16 + m;
    float bo = bout[n];
    float mu = mean[n];
    float rv = rsqrtf(var[n] + 1e-5f);
    float g = gamma[n], be = beta[n];
    if (valid) {
#pragma unroll
      for (int v = 0; v < 8; v++) {
        int nm = t * 16 + mbase + v;
        float x = c[v] + bo + h32[(size_t)nm * HID_ + n];  // residual (dst)
        x = (x - mu) * rv * g + be;
        x = selu_f(x);
        h32[(size_t)nm * HID_ + n] = x;
        h16[(size_t)nm * HID_ + n] = (_Float16)x;
        if (nm < N2_) interBuf[(size_t)nm * HID_ + n] = x;
      }
    }
  }
}

// ------------ proj_out layer 1: concat(i0,i1,i2)(384) @ W1 -> selu ----------
__global__ __launch_bounds__(256) void proj_out1_kernel(
    const float* __restrict__ i0, const float* __restrict__ i1,
    const float* __restrict__ i2,
    const _Float16* __restrict__ w1p, const float* __restrict__ b1,
    float* __restrict__ zhid) {
  const int lane = threadIdx.x & 31;
  const int wave = threadIdx.x >> 5;
  const int tile = blockIdx.x * 8 + wave;
  const int nTiles = N2_ / 16;
  const bool valid = tile < nTiles;
  const int t = valid ? tile : 0;
  const int m = lane & 15;
  const int mbase = (lane >> 4) << 3;
  const int node = t * 16 + m;

  v16h A[12];
#pragma unroll
  for (int ks = 0; ks < 4; ks++) {
    A[ks]     = gatherA(i0 + (size_t)node * HID_, ks, lane);
    A[4 + ks] = gatherA(i1 + (size_t)node * HID_, ks, lane);
    A[8 + ks] = gatherA(i2 + (size_t)node * HID_, ks, lane);
  }
#pragma unroll
  for (int nt = 0; nt < 8; nt++) {
    v8f c = {};
#pragma unroll
    for (int ks = 0; ks < 12; ks++) c = wmma16(A[ks], ldb(w1p, nt, 12, ks, lane), c);
    int n = nt * 16 + m;
    float bias = b1[n];
    if (valid) {
#pragma unroll
      for (int v = 0; v < 8; v++) {
        int nm = t * 16 + mbase + v;
        zhid[(size_t)nm * HID_ + n] = selu_f(c[v] + bias);
      }
    }
  }
}

// ------------ final head: 128->2 + log_softmax -----------------------------
__global__ void final_kernel(const float* __restrict__ zhid,
                             const float* __restrict__ W2,
                             const float* __restrict__ b2,
                             float* __restrict__ out) {
  int node = blockIdx.x * blockDim.x + threadIdx.x;
  if (node >= N2_) return;
  const float* z = zhid + (size_t)node * HID_;
  float y0 = b2[0], y1 = b2[1];
#pragma unroll 8
  for (int k = 0; k < HID_; k++) {
    float zv = z[k];
    y0 += zv * W2[k * 2 + 0];
    y1 += zv * W2[k * 2 + 1];
  }
  float mx = fmaxf(y0, y1);
  float l = logf(__expf(y0 - mx) + __expf(y1 - mx));
  out[node * 2 + 0] = y0 - mx - l;
  out[node * 2 + 1] = y1 - mx - l;
}

// ------------------------- workspace layout --------------------------------
static constexpr size_t HV = sizeof(_Float16);
static constexpr size_t SZ_H32 = (size_t)N0_ * HID_ * sizeof(float);
static constexpr size_t SZ_H16 = (size_t)N0_ * HID_ * HV;
static constexpr size_t SZ_AGG = (size_t)N1_ * HID_ * sizeof(float);
static constexpr size_t SZ_I   = (size_t)N2_ * HID_ * sizeof(float);
static constexpr size_t OFF_H32 = 0;
static constexpr size_t OFF_H16 = OFF_H32 + SZ_H32;
static constexpr size_t OFF_AGG = OFF_H16 + SZ_H16;
static constexpr size_t OFF_I0  = OFF_AGG + SZ_AGG;
static constexpr size_t OFF_I1  = OFF_I0 + SZ_I;
static constexpr size_t OFF_I2  = OFF_I1 + SZ_I;
static constexpr size_t OFF_ZH  = OFF_I2 + SZ_I;
static constexpr size_t OFF_PIW1  = OFF_ZH + SZ_I;
static constexpr size_t OFF_PIW2  = OFF_PIW1 + (size_t)64 * 128 * HV;
static constexpr size_t OFF_WOUT0 = OFF_PIW2 + (size_t)128 * 128 * HV;
static constexpr size_t OFF_WOUT1 = OFF_WOUT0 + (size_t)128 * 128 * HV;
static constexpr size_t OFF_EW    = OFF_WOUT1 + (size_t)128 * 128 * HV;
static constexpr size_t EW_STRIDE = ((size_t)256 * 64 + (size_t)64 * 128) * HV;
static constexpr size_t OFF_POW1  = OFF_EW + 6 * EW_STRIDE;

extern "C" void kernel_launch(void* const* d_in, const int* in_sizes, int n_in,
                              void* d_out, int out_size, void* d_ws, size_t ws_size,
                              hipStream_t stream) {
  (void)in_sizes; (void)n_in; (void)out_size; (void)ws_size;
  const float* feat = (const float*)d_in[0];
  const int* esrc[2][3];
  const int* edst[2][3];
  for (int l = 0; l < 2; l++)
    for (int e = 0; e < 3; e++) {
      esrc[l][e] = (const int*)d_in[1 + l * 6 + e * 2];
      edst[l][e] = (const int*)d_in[2 + l * 6 + e * 2];
    }

  char* ws = (char*)d_ws;
  float*    h32  = (float*)(ws + OFF_H32);
  _Float16* h16  = (_Float16*)(ws + OFF_H16);
  float*    agg  = (float*)(ws + OFF_AGG);
  float*    i0   = (float*)(ws + OFF_I0);
  float*    i1   = (float*)(ws + OFF_I1);
  float*    i2   = (float*)(ws + OFF_I2);
  float*    zhid = (float*)(ws + OFF_ZH);
  _Float16* piW1p  = (_Float16*)(ws + OFF_PIW1);
  _Float16* piW2p  = (_Float16*)(ws + OFF_PIW2);
  _Float16* wout0p = (_Float16*)(ws + OFF_WOUT0);
  _Float16* wout1p = (_Float16*)(ws + OFF_WOUT1);
  _Float16* poW1p  = (_Float16*)(ws + OFF_POW1);

  auto pack = [&](const void* W, _Float16* outp, int K, int N) {
    int tot = K * N;
    pack_b_kernel<<<(tot + 255) / 256, 256, 0, stream>>>((const float*)W, outp, K, N);
  };

  // pack all weight matrices into B-fragment order (f16)
  pack(d_in[49], piW1p, 64, 128);     // proj_in.W1
  pack(d_in[50], piW2p, 128, 128);    // proj_in.W2
  pack(d_in[13], wout0p, 128, 128);   // layers[0].Wout
  pack(d_in[31], wout1p, 128, 128);   // layers[1].Wout
  _Float16* eW1p[2][3];
  _Float16* eW2p[2][3];
  for (int l = 0; l < 2; l++) {
    int base = 13 + 18 * l;
    for (int e = 0; e < 3; e++) {
      char* blk = ws + OFF_EW + ((size_t)(l * 3 + e)) * EW_STRIDE;
      eW1p[l][e] = (_Float16*)blk;
      eW2p[l][e] = (_Float16*)(blk + (size_t)256 * 64 * HV);
      pack(d_in[base + 6 + 4 * e], eW1p[l][e], 256, 64);   // em.W1
      pack(d_in[base + 7 + 4 * e], eW2p[l][e], 64, 128);   // em.W2
    }
  }
  pack(d_in[53], poW1p, 384, 128);    // proj_out.W1

  // proj_in over all N0 nodes
  {
    int nTiles = N0_ / 16;
    proj_in_kernel<<<(nTiles + 7) / 8, 256, 0, stream>>>(
        feat, piW1p, (const float*)d_in[51], piW2p, (const float*)d_in[52],
        h32, h16, i0);
  }

  // two message-passing layers
  for (int l = 0; l < 2; l++) {
    const int numDst = (l == 0) ? N1_ : N2_;
    const int E = (l == 0) ? E0_ : E1_;
    const int base = 13 + 18 * l;
    const int aggN = numDst * HID_;
    zero_kernel<<<(aggN + 255) / 256, 256, 0, stream>>>(agg, aggN);
    for (int e = 0; e < 3; e++) {
      int nTiles32 = E / 32;
      edge_mlp_kernel<<<(nTiles32 + 7) / 8, 256, 0, stream>>>(
          h16, esrc[l][e], edst[l][e],
          eW1p[l][e], (const float*)d_in[base + 8 + 4 * e],
          eW2p[l][e], (const float*)d_in[base + 9 + 4 * e],
          agg, nTiles32);
    }
    int nTiles = numDst / 16;
    node_update_kernel<<<(nTiles + 7) / 8, 256, 0, stream>>>(
        agg, (l == 0 ? wout0p : wout1p), (const float*)d_in[base + 5],
        (const float*)d_in[base + 1], (const float*)d_in[base + 2],
        (const float*)d_in[base + 3], (const float*)d_in[base + 4],
        h32, h16, (l == 0 ? i1 : i2), nTiles);
  }

  // output head
  {
    int nTiles = N2_ / 16;
    proj_out1_kernel<<<(nTiles + 7) / 8, 256, 0, stream>>>(
        i0, i1, i2, poW1p, (const float*)d_in[55], zhid);
    final_kernel<<<(N2_ + 255) / 256, 256, 0, stream>>>(
        zhid, (const float*)d_in[54], (const float*)d_in[56], (float*)d_out);
  }
}
